// tDBN_1_12841952215182
// MI455X (gfx1250) — matrix operand order repaired and checked by reading, not compiled, not run on hardware
//
#include <hip/hip_runtime.h>
#include <stdint.h>
#include <stddef.h>

// ---------------------------------------------------------------------------
// CDNA5 / gfx1250 WMMA types
// ---------------------------------------------------------------------------
typedef __bf16 bf16;
typedef __attribute__((ext_vector_type(8)))  __bf16 v8bf;
typedef __attribute__((ext_vector_type(16))) __bf16 v16bf;
typedef __attribute__((ext_vector_type(8)))  float  v8f;

// ---------------------------------------------------------------------------
// Utility kernels
// ---------------------------------------------------------------------------
__global__ void zero_u32(uint32_t* __restrict__ p, size_t n) {
    size_t i  = (size_t)blockIdx.x * blockDim.x + threadIdx.x;
    size_t st = (size_t)gridDim.x * blockDim.x;
    for (; i < n; i += st) p[i] = 0u;
}

// pass 1: deterministic winner per voxel site (max voxel index wins)
__global__ void scatter_win(const int* __restrict__ coors, int* __restrict__ winner,
                            int nvox, int H, int W) {
    int v = blockIdx.x * blockDim.x + threadIdx.x;
    if (v >= nvox) return;
    int z = coors[v * 4 + 1], y = coors[v * 4 + 2], x = coors[v * 4 + 3];
    size_t site = ((size_t)z * H + y) * W + x;
    atomicMax(&winner[site], v + 1);
}

// pass 2: winner writes its features (bf16, NDHWC) and the active mask
__global__ void scatter_feat(const float* __restrict__ vf, const int* __restrict__ coors,
                             const int* __restrict__ winner,
                             bf16* __restrict__ xd, uint8_t* __restrict__ m0,
                             int nvox, int Cin, int H, int W) {
    int i = blockIdx.x * blockDim.x + threadIdx.x;
    int v = i / Cin, c = i % Cin;
    if (v >= nvox) return;
    int z = coors[v * 4 + 1], y = coors[v * 4 + 2], x = coors[v * 4 + 3];
    size_t site = ((size_t)z * H + y) * W + x;
    if (winner[site] != v + 1) return;
    xd[site * Cin + c] = (bf16)vf[(size_t)v * Cin + c];
    if (c == 0) m0[site] = 1;
}

// active-site propagation of a VALID-padding SparseConv3d
__global__ void dilate_k(const uint8_t* __restrict__ mi, uint8_t* __restrict__ mo,
                         int Hin, int Win,
                         int Do, int Ho, int Wo,
                         int KZ, int KH, int KW, int SZ, int SH, int SW) {
    int s  = blockIdx.x * blockDim.x + threadIdx.x;
    int nS = Do * Ho * Wo;
    if (s >= nS) return;
    int w = s % Wo, h = (s / Wo) % Ho, d = s / (Wo * Ho);
    int acc = 0;
    for (int kz = 0; kz < KZ; ++kz)
        for (int ky = 0; ky < KH; ++ky)
            for (int kx = 0; kx < KW; ++kx) {
                int zi = d * SZ + kz, yi = h * SH + ky, xi = w * SW + kx;
                acc |= mi[((size_t)zi * Hin + yi) * Win + xi];
            }
    mo[s] = acc ? 1 : 0;
}

// compact active-site index list from a mask (order-independent results)
__global__ void build_list(const uint8_t* __restrict__ m, int* __restrict__ list,
                           int* __restrict__ cnt, int nS) {
    int s = blockIdx.x * blockDim.x + threadIdx.x;
    if (s >= nS) return;
    if (m[s]) {
        int p = atomicAdd(cnt, 1);
        list[p] = s;
    }
}

// Pre-pack OIDHW f32 weights into per-lane B-fragment order (bf16).
// conv reads wf[(((n*T + t)*Q + q)*32 + lane)*16 + j]
//   == W[co = n*16 + (lane&15)][ci = q*32 + (lane>>4)*8 + (j&7) + (j>>3)*16][tap t]
__global__ void wprep_k(const float* __restrict__ w, bf16* __restrict__ wf,
                        int Cin, int T, int Q, int NT) {
    size_t idx   = (size_t)blockIdx.x * blockDim.x + threadIdx.x;
    size_t total = (size_t)NT * T * Q * 512;
    if (idx >= total) return;
    int    j    = (int)(idx & 15);
    int    lane = (int)((idx >> 4) & 31);
    size_t rest = idx >> 9;
    int    q    = (int)(rest % Q); rest /= Q;
    int    t    = (int)(rest % T);
    int    n    = (int)(rest / T);
    int co = n * 16 + (lane & 15);
    int ci = q * 32 + ((lane >> 4) << 3) + (j & 7) + ((j >> 3) << 4);
    wf[idx] = (bf16)w[((size_t)co * Cin + ci) * T + t];
}

// ---------------------------------------------------------------------------
// Sparse implicit-GEMM conv with V_WMMA_F32_16X16X32_BF16 + fused BN/ReLU.
// M tiles are gathered from the active-site list; inactive sites stay zero
// (output buffers are pre-zeroed), which reproduces the reference mask.
// B fragments are reused across MT tiles; A fragments across NT co-blocks.
// ---------------------------------------------------------------------------
template <int NT, int MT, int Q, bool PAD, bool F32OUT>
__global__ __launch_bounds__(128) void conv_wmma(
    const bf16* __restrict__ in, const bf16* __restrict__ wf,
    const float* __restrict__ bns, const float* __restrict__ bnb,
    const int* __restrict__ list, const int* __restrict__ cntp,
    bf16* __restrict__ outb, float* __restrict__ outf,
    int Din, int Hin, int Win,
    int Do, int Ho, int Wo,
    int KZ, int KH, int KW, int SZ, int SH, int SW) {
    const int Cin  = Q * 32;
    const int T    = KZ * KH * KW;
    const int nS   = Do * Ho * Wo;
    const int cnt  = *cntp;  // wave-uniform scalar load
    const int wave = blockIdx.x * (blockDim.x >> 5) + (threadIdx.x >> 5);
    const int base0 = wave * (16 * MT);
    if (base0 >= cnt) return;  // wave-uniform: EXEC stays all-ones for WMMA
    const int lane = threadIdx.x & 31;

    // A-fragment row (M = lane&15); ISA 16-bit A layout: lane holds
    // K = {kb..kb+7, kb+16..kb+23}, kb = (lane>=16)*8
    const int kbase = (lane >> 4) << 3;
    int dA[MT], hA[MT], wA[MT];
#pragma unroll
    for (int mt = 0; mt < MT; ++mt) {
        int idx = base0 + mt * 16 + (lane & 15);
        if (idx >= cnt) idx = cnt - 1;
        const int sA = list[idx];
        wA[mt] = sA % Wo;
        hA[mt] = (sA / Wo) % Ho;
        dA[mt] = sA / (Wo * Ho);
    }

    const v8f zf = {};
    v8f acc[MT * NT];
#pragma unroll
    for (int i = 0; i < MT * NT; ++i) acc[i] = zf;

    const size_t nStride = (size_t)T * Q * 512;  // halfs per co-block in wf

    for (int t = 0; t < T; ++t) {
        const int kx = t % KW;
        const int ky = (t / KW) % KH;
        const int kz = t / (KW * KH);

        const bf16* ap0[MT];
        bool ok[MT];
#pragma unroll
        for (int mt = 0; mt < MT; ++mt) {
            const int zi = dA[mt] * SZ + kz - (PAD ? 1 : 0);
            const int yi = hA[mt] * SH + ky - (PAD ? 1 : 0);
            const int xi = wA[mt] * SW + kx - (PAD ? 1 : 0);
            ok[mt] = true;
            if (PAD)
                ok[mt] = (unsigned)zi < (unsigned)Din &&
                         (unsigned)yi < (unsigned)Hin &&
                         (unsigned)xi < (unsigned)Win;
            ap0[mt] = ok[mt]
                          ? (in + (((size_t)zi * Hin + yi) * Win + xi) * Cin + kbase)
                          : in;
        }

#pragma unroll
        for (int q = 0; q < Q; ++q) {
            v16bf a[MT];
#pragma unroll
            for (int mt = 0; mt < MT; ++mt) {
                a[mt] = (v16bf){};
                if (!PAD || ok[mt]) {
                    const v8bf* ap = (const v8bf*)(ap0[mt] + q * 32);
                    v8bf lo = ap[0];  // K: kbase .. kbase+7
                    v8bf hi = ap[2];  // K: kbase+16 .. kbase+23
                    a[mt] = __builtin_shufflevector(lo, hi, 0, 1, 2, 3, 4, 5, 6,
                                                    7, 8, 9, 10, 11, 12, 13, 14,
                                                    15);
                }
            }
            const bf16* wp = wf + (((size_t)t * Q + q) * 32 + lane) * 16;
#pragma unroll
            for (int n = 0; n < NT; ++n) {
                const v8bf* bp = (const v8bf*)(wp + (size_t)n * nStride);
                v16bf bm = __builtin_shufflevector(bp[0], bp[1], 0, 1, 2, 3, 4, 5,
                                                   6, 7, 8, 9, 10, 11, 12, 13, 14,
                                                   15);
#pragma unroll
                for (int mt = 0; mt < MT; ++mt) {
                    acc[mt * NT + n] = __builtin_amdgcn_wmma_f32_16x16x32_bf16(
                        false, a[mt], false, bm, (short)0, acc[mt * NT + n],
                        false, false);
                }
            }
        }
    }

    // Epilogue: C/D layout M = (lane>=16)*8 + r, N = lane&15
    const int col   = lane & 15;
    const int rbase = (lane >> 4) << 3;
#pragma unroll
    for (int mt = 0; mt < MT; ++mt) {
#pragma unroll
        for (int r = 0; r < 8; ++r) {
            const int idx = base0 + mt * 16 + rbase + r;
            if (idx < cnt) {
                const int s = list[idx];
#pragma unroll
                for (int n = 0; n < NT; ++n) {
                    const int co = n * 16 + col;
                    float v = acc[mt * NT + n][r] * bns[co] + bnb[co];
                    v = (v > 0.0f ? v : 0.0f);
                    if (F32OUT)
                        outf[(size_t)co * nS + s] = v;           // NCDHW flat
                    else
                        outb[(size_t)s * (NT * 16) + co] = (bf16)v;  // NDHWC bf16
                }
            }
        }
    }
}

// ---------------------------------------------------------------------------
// Host-side dispatch (worst-case grid; waves early-exit on the active count)
// ---------------------------------------------------------------------------
template <int NT, int MT, int Q, bool PAD, bool F32OUT>
static void conv_launch(hipStream_t st, const bf16* in, const bf16* wf,
                        const float* s, const float* b, const int* list,
                        const int* cnt, bf16* outb, float* outf, int Din,
                        int Hin, int Win, int Do, int Ho, int Wo, int KZ,
                        int KH, int KW, int SZ, int SH, int SW) {
    int nS = Do * Ho * Wo;
    int tiles16 = (nS + 15) / 16;
    int waves = (tiles16 + MT - 1) / MT;
    int blocks = (waves + 3) / 4;  // 4 waves / block (128 threads, wave32)
    conv_wmma<NT, MT, Q, PAD, F32OUT><<<blocks, 128, 0, st>>>(
        in, wf, s, b, list, cnt, outb, outf, Din, Hin, Win, Do, Ho, Wo, KZ, KH,
        KW, SZ, SH, SW);
}

static void wprep_launch(hipStream_t st, const float* w, bf16* wf, int Cin,
                         int T, int Q, int NT) {
    size_t total = (size_t)NT * T * Q * 512;
    int blocks = (int)((total + 255) / 256);
    wprep_k<<<blocks, 256, 0, st>>>(w, wf, Cin, T, Q, NT);
}

extern "C" void kernel_launch(void* const* d_in, const int* in_sizes, int n_in,
                              void* d_out, int out_size, void* d_ws,
                              size_t ws_size, hipStream_t stream) {
    (void)n_in; (void)out_size; (void)ws_size;
    // ---- inputs (flattened setup_inputs() dict order) ----
    const float* vf    = (const float*)d_in[0];
    const int*   coors = (const int*)d_in[1];
    const int nvox = in_sizes[0] / 128;
    const float *b0_w = (const float*)d_in[3],  *b0_s = (const float*)d_in[4],  *b0_b = (const float*)d_in[5];
    const float *fm0_w = (const float*)d_in[6], *fm0_s = (const float*)d_in[7], *fm0_b = (const float*)d_in[8];
    const float *b1_dw = (const float*)d_in[9], *b1_ds = (const float*)d_in[10], *b1_db = (const float*)d_in[11];
    const float *b1_w1 = (const float*)d_in[12], *b1_s1 = (const float*)d_in[13], *b1_b1 = (const float*)d_in[14];
    const float *b1_w2 = (const float*)d_in[15], *b1_s2 = (const float*)d_in[16], *b1_b2 = (const float*)d_in[17];
    const float *b2_dw = (const float*)d_in[18], *b2_ds = (const float*)d_in[19], *b2_db = (const float*)d_in[20];
    const float *b2_w1 = (const float*)d_in[21], *b2_s1 = (const float*)d_in[22], *b2_b1 = (const float*)d_in[23];
    const float *b2_w2 = (const float*)d_in[24], *b2_s2 = (const float*)d_in[25], *b2_b2 = (const float*)d_in[26];
    const float *b3_dw = (const float*)d_in[27], *b3_ds = (const float*)d_in[28], *b3_db = (const float*)d_in[29];
    const float *b3_w1 = (const float*)d_in[30], *b3_s1 = (const float*)d_in[31], *b3_b1 = (const float*)d_in[32];
    const float *b3_w2 = (const float*)d_in[33], *b3_s2 = (const float*)d_in[34], *b3_b2 = (const float*)d_in[35];
    const float *fm1_w = (const float*)d_in[36], *fm1_s = (const float*)d_in[37], *fm1_b = (const float*)d_in[38];
    const float *fm2_w = (const float*)d_in[39], *fm2_s = (const float*)d_in[40], *fm2_b = (const float*)d_in[41];

    // ---- grid geometry ----
    const int D0 = 17, H0 = 128, W0 = 128;
    const size_t S0 = (size_t)D0 * H0 * W0;          // 278528
    const int D1 = 8, H1 = 63, W1 = 63;  const size_t S1 = (size_t)D1 * H1 * W1;   // 31752
    const int D2 = 3, H2 = 31, W2 = 31;  const size_t S2 = (size_t)D2 * H2 * W2;   // 2883
    const int D3 = 1, H3 = 15, W3 = 15;  const size_t S3 = (size_t)D3 * H3 * W3;   // 225
    const size_t SY0 = (size_t)2 * 128 * 128, SY1 = (size_t)63 * 63, SY2 = (size_t)31 * 31;

    // ---- workspace bump allocator ----
    size_t off = 0;
    auto alloc = [&](size_t bytes) -> char* {
        char* p = (char*)d_ws + off;
        off += (bytes + 255) & ~(size_t)255;
        return p;
    };
    bf16*    xdense = (bf16*)alloc(S0 * 128 * sizeof(bf16));
    uint8_t* m0     = (uint8_t*)alloc(S0);
    int*     winner = (int*)alloc(S0 * sizeof(int));
    bf16*    x0     = (bf16*)alloc(S0 * 32 * sizeof(bf16));
    bf16*    x1a    = (bf16*)alloc(S1 * 64 * sizeof(bf16));
    bf16*    x1b    = (bf16*)alloc(S1 * 64 * sizeof(bf16));
    bf16*    x2a    = (bf16*)alloc(S2 * 96 * sizeof(bf16));
    bf16*    x2b    = (bf16*)alloc(S2 * 96 * sizeof(bf16));
    bf16*    x3a    = (bf16*)alloc(S3 * 128 * sizeof(bf16));
    bf16*    x3b    = (bf16*)alloc(S3 * 128 * sizeof(bf16));
    uint8_t* m1  = (uint8_t*)alloc(S1);
    uint8_t* m2  = (uint8_t*)alloc(S2);
    uint8_t* m3  = (uint8_t*)alloc(S3);
    uint8_t* my0 = (uint8_t*)alloc(SY0);
    uint8_t* my1 = (uint8_t*)alloc(SY1);
    uint8_t* my2 = (uint8_t*)alloc(SY2);
    // active-site lists + counters
    int* l_m0  = (int*)alloc(S0 * sizeof(int));
    int* l_m1  = (int*)alloc(S1 * sizeof(int));
    int* l_m2  = (int*)alloc(S2 * sizeof(int));
    int* l_m3  = (int*)alloc(S3 * sizeof(int));
    int* l_my0 = (int*)alloc(SY0 * sizeof(int));
    int* l_my1 = (int*)alloc(SY1 * sizeof(int));
    int* l_my2 = (int*)alloc(SY2 * sizeof(int));
    int* cnts  = (int*)alloc(8 * sizeof(int));  // [m0,m1,m2,m3,my0,my1,my2]
    auto walloc = [&](int NT, int T, int Q) -> bf16* {
        return (bf16*)alloc((size_t)NT * T * Q * 512 * sizeof(bf16));
    };
    bf16* wf_b0  = walloc(2, 27, 4);
    bf16* wf_fm0 = walloc(8, 15, 1);
    bf16* wf_b1d = walloc(4, 27, 1);
    bf16* wf_b11 = walloc(4, 27, 2);
    bf16* wf_b12 = walloc(4, 27, 2);
    bf16* wf_fm1 = walloc(8, 7, 2);
    bf16* wf_b2d = walloc(6, 27, 2);
    bf16* wf_b21 = walloc(6, 27, 3);
    bf16* wf_b22 = walloc(6, 27, 3);
    bf16* wf_fm2 = walloc(8, 3, 3);
    bf16* wf_b3d = walloc(8, 27, 3);
    bf16* wf_b31 = walloc(8, 27, 4);
    bf16* wf_b32 = walloc(8, 27, 4);

    float* outF = (float*)d_out;
    const size_t Y0_OFF = 0, Y1_OFF = 4194304, Y2_OFF = 4702336, X3_OFF = 4825344;
    const size_t OUT_TOTAL = 4854144;

    // ---- stage 0: zero everything that must start at zero ----
    zero_u32<<<2048, 256, 0, stream>>>((uint32_t*)xdense, S0 * 128 * 2 / 4);
    zero_u32<<<256, 256, 0, stream>>>((uint32_t*)m0, S0 / 4);
    zero_u32<<<512, 256, 0, stream>>>((uint32_t*)winner, S0);
    zero_u32<<<1024, 256, 0, stream>>>((uint32_t*)x0, S0 * 32 * 2 / 4);
    zero_u32<<<256, 256, 0, stream>>>((uint32_t*)x1a, S1 * 64 * 2 / 4);
    zero_u32<<<256, 256, 0, stream>>>((uint32_t*)x1b, S1 * 64 * 2 / 4);
    zero_u32<<<64, 256, 0, stream>>>((uint32_t*)x2a, S2 * 96 * 2 / 4);
    zero_u32<<<64, 256, 0, stream>>>((uint32_t*)x2b, S2 * 96 * 2 / 4);
    zero_u32<<<16, 256, 0, stream>>>((uint32_t*)x3a, S3 * 128 * 2 / 4);
    zero_u32<<<16, 256, 0, stream>>>((uint32_t*)x3b, S3 * 128 * 2 / 4);
    zero_u32<<<1024, 256, 0, stream>>>((uint32_t*)outF, OUT_TOTAL);
    zero_u32<<<1, 32, 0, stream>>>((uint32_t*)cnts, 8);

    // ---- scatter voxels ----
    scatter_win<<<(nvox + 255) / 256, 256, 0, stream>>>(coors, winner, nvox, H0, W0);
    scatter_feat<<<(nvox * 128 + 255) / 256, 256, 0, stream>>>(
        vf, coors, winner, xdense, m0, nvox, 128, H0, W0);

    // ---- weight pre-pack (f32 OIDHW -> bf16 B-fragment order) ----
    wprep_launch(stream, b0_w,  wf_b0,  128, 27, 4, 2);
    wprep_launch(stream, fm0_w, wf_fm0, 32,  15, 1, 8);
    wprep_launch(stream, b1_dw, wf_b1d, 32,  27, 1, 4);
    wprep_launch(stream, b1_w1, wf_b11, 64,  27, 2, 4);
    wprep_launch(stream, b1_w2, wf_b12, 64,  27, 2, 4);
    wprep_launch(stream, fm1_w, wf_fm1, 64,  7,  2, 8);
    wprep_launch(stream, b2_dw, wf_b2d, 64,  27, 2, 6);
    wprep_launch(stream, b2_w1, wf_b21, 96,  27, 3, 6);
    wprep_launch(stream, b2_w2, wf_b22, 96,  27, 3, 6);
    wprep_launch(stream, fm2_w, wf_fm2, 96,  3,  3, 8);
    wprep_launch(stream, b3_dw, wf_b3d, 96,  27, 3, 8);
    wprep_launch(stream, b3_w1, wf_b31, 128, 27, 4, 8);
    wprep_launch(stream, b3_w2, wf_b32, 128, 27, 4, 8);

    // ---- active-set dilation masks ----
    dilate_k<<<(int)((S1 + 255) / 256), 256, 0, stream>>>(m0, m1, H0, W0, D1, H1, W1, 3, 3, 3, 2, 2, 2);
    dilate_k<<<(int)((SY0 + 255) / 256), 256, 0, stream>>>(m0, my0, H0, W0, 2, 128, 128, 15, 1, 1, 2, 1, 1);
    dilate_k<<<(int)((S2 + 255) / 256), 256, 0, stream>>>(m1, m2, H1, W1, D2, H2, W2, 3, 3, 3, 2, 2, 2);
    dilate_k<<<(int)((SY1 + 255) / 256), 256, 0, stream>>>(m1, my1, H1, W1, 1, 63, 63, 7, 1, 1, 2, 1, 1);
    dilate_k<<<(int)((S3 + 255) / 256), 256, 0, stream>>>(m2, m3, H2, W2, D3, H3, W3, 3, 3, 3, 2, 2, 2);
    dilate_k<<<(int)((SY2 + 255) / 256), 256, 0, stream>>>(m2, my2, H2, W2, 1, 31, 31, 3, 1, 1, 2, 1, 1);

    // ---- compact active-site lists ----
    build_list<<<(int)((S0 + 255) / 256), 256, 0, stream>>>(m0, l_m0, cnts + 0, (int)S0);
    build_list<<<(int)((S1 + 255) / 256), 256, 0, stream>>>(m1, l_m1, cnts + 1, (int)S1);
    build_list<<<(int)((S2 + 255) / 256), 256, 0, stream>>>(m2, l_m2, cnts + 2, (int)S2);
    build_list<<<(int)((S3 + 255) / 256), 256, 0, stream>>>(m3, l_m3, cnts + 3, (int)S3);
    build_list<<<(int)((SY0 + 255) / 256), 256, 0, stream>>>(my0, l_my0, cnts + 4, (int)SY0);
    build_list<<<(int)((SY1 + 255) / 256), 256, 0, stream>>>(my1, l_my1, cnts + 5, (int)SY1);
    build_list<<<(int)((SY2 + 255) / 256), 256, 0, stream>>>(my2, l_my2, cnts + 6, (int)SY2);

    // ---- network (active sites only; inactive stay zero) ----
    // b0: SubM 128->32, pad1
    conv_launch<2, 4, 4, true, false>(stream, xdense, wf_b0, b0_s, b0_b, l_m0,
                                      cnts + 0, x0, nullptr, D0, H0, W0, D0, H0,
                                      W0, 3, 3, 3, 1, 1, 1);
    // fm0: SparseConv 32->128, k(15,1,1) s(2,1,1) -> y0 (f32, NCDHW flat)
    conv_launch<8, 2, 1, false, true>(stream, x0, wf_fm0, fm0_s, fm0_b, l_my0,
                                      cnts + 4, nullptr, outF + Y0_OFF, D0, H0,
                                      W0, 2, 128, 128, 15, 1, 1, 2, 1, 1);
    // b1
    conv_launch<4, 4, 1, false, false>(stream, x0, wf_b1d, b1_ds, b1_db, l_m1,
                                       cnts + 1, x1a, nullptr, D0, H0, W0, D1,
                                       H1, W1, 3, 3, 3, 2, 2, 2);
    conv_launch<4, 4, 2, true, false>(stream, x1a, wf_b11, b1_s1, b1_b1, l_m1,
                                      cnts + 1, x1b, nullptr, D1, H1, W1, D1,
                                      H1, W1, 3, 3, 3, 1, 1, 1);
    conv_launch<4, 4, 2, true, false>(stream, x1b, wf_b12, b1_s2, b1_b2, l_m1,
                                      cnts + 1, x1a, nullptr, D1, H1, W1, D1,
                                      H1, W1, 3, 3, 3, 1, 1, 1);
    conv_launch<8, 2, 2, false, true>(stream, x1a, wf_fm1, fm1_s, fm1_b, l_my1,
                                      cnts + 5, nullptr, outF + Y1_OFF, D1, H1,
                                      W1, 1, 63, 63, 7, 1, 1, 2, 1, 1);
    // b2
    conv_launch<6, 2, 2, false, false>(stream, x1a, wf_b2d, b2_ds, b2_db, l_m2,
                                       cnts + 2, x2a, nullptr, D1, H1, W1, D2,
                                       H2, W2, 3, 3, 3, 2, 2, 2);
    conv_launch<6, 2, 3, true, false>(stream, x2a, wf_b21, b2_s1, b2_b1, l_m2,
                                      cnts + 2, x2b, nullptr, D2, H2, W2, D2,
                                      H2, W2, 3, 3, 3, 1, 1, 1);
    conv_launch<6, 2, 3, true, false>(stream, x2b, wf_b22, b2_s2, b2_b2, l_m2,
                                      cnts + 2, x2a, nullptr, D2, H2, W2, D2,
                                      H2, W2, 3, 3, 3, 1, 1, 1);
    conv_launch<8, 2, 3, false, true>(stream, x2a, wf_fm2, fm2_s, fm2_b, l_my2,
                                      cnts + 6, nullptr, outF + Y2_OFF, D2, H2,
                                      W2, 1, 31, 31, 3, 1, 1, 2, 1, 1);
    // b3
    conv_launch<8, 2, 3, false, false>(stream, x2a, wf_b3d, b3_ds, b3_db, l_m3,
                                       cnts + 3, x3a, nullptr, D2, H2, W2, D3,
                                       H3, W3, 3, 3, 3, 2, 2, 2);
    conv_launch<8, 2, 4, true, false>(stream, x3a, wf_b31, b3_s1, b3_b1, l_m3,
                                      cnts + 3, x3b, nullptr, D3, H3, W3, D3,
                                      H3, W3, 3, 3, 3, 1, 1, 1);
    conv_launch<8, 2, 4, true, true>(stream, x3b, wf_b32, b3_s2, b3_b2, l_m3,
                                     cnts + 3, nullptr, outF + X3_OFF, D3, H3,
                                     W3, D3, H3, W3, 3, 3, 3, 1, 1, 1);
}
